// DecoderWithAttention_71287867179649
// MI455X (gfx1250) — matrix operand order repaired and checked by reading, not compiled, not run on hardware
//
#include <hip/hip_runtime.h>

// ---------------------------------------------------------------------------
// Bottom-Up/Top-Down caption decoder forward pass for MI455X (gfx1250).
//
// Strategy (reasoned from MI455X specs, no runtime measurement possible):
//  * Weights (W_ih1|W_hh1, W_ih2|W_hh2, Wd, Wfc1, Wfc, We) converted once to
//    bf16 (~119 MB) -> entire weight set is L2-resident (192 MB L2), so the 19
//    sequential timesteps re-read weights from L2 instead of HBM.
//  * All GEMMs use v_wmma_f32_16x16x32_bf16 (fp32 accumulate). M=64 batch ->
//    4 WMMA M-tiles; every K is a multiple of 64; every N a multiple of 16.
//  * GEMM inner loop: double-buffered LDS (one barrier per 64-K chunk),
//    software-pipelined next-chunk global loads, global_prefetch on the W
//    stream, and ALL 8 A-fragments loaded before the 8 WMMAs (enforced with
//    __builtin_amdgcn_sched_barrier) so the WMMAs issue back-to-back.
//  * att1 (B,P,ATT) precomputed once via a real 12544x1024x2048 WMMA GEMM and
//    kept in bf16 (25.7 MB) to minimize per-step attention traffic.
//  * LSTM concats folded into concatenated weights so each LSTM input path is
//    ONE GEMM:  gates = [x | h] @ [W_ih | W_hh]^T.
//
// Workspace requirement: ~206 MB.
// ---------------------------------------------------------------------------

#define B_   64
#define P_   196
#define ENC_ 2048
#define DEC_ 1024
#define ATT_ 1024
#define EMB_ 1024
#define V_   10000
#define T_   19      // Tdec = MAXLEN - 1
#define ML_  20

typedef __attribute__((ext_vector_type(16))) __bf16 v16bf;
typedef __attribute__((ext_vector_type(8)))  float  v8f;

#define LDA_PAD 72   // LDS row stride in bf16 elems (144 B = 9*16B, conflict-free)

// ---------------------------------------------------------------------------
// Generic bf16 WMMA GEMM:  out[m,n] = sum_k A[m,k]*W[n,k] (+ bias[n]), with
// optional per-row masking (t >= dec_len[m] -> 0) and fp32 or bf16 output.
// Block = 256 threads = 8 waves. Block tile: 64(M) x 128(N). grid.y tiles M.
// ---------------------------------------------------------------------------
__global__ __launch_bounds__(256)
void gemm_bf16_wmma(const __bf16* __restrict__ A, int K,
                    const __bf16* __restrict__ W, int N,
                    const float* __restrict__ bias,
                    void* __restrict__ out, long ostride, int out_bf16,
                    const int* __restrict__ dec_len, int t)
{
    __shared__ __bf16 sA[2][64 * LDA_PAD];

    const int tid  = threadIdx.x;
    const int wave = tid >> 5;
    const int lane = tid & 31;
    const int l15  = lane & 15;
    const int mbase = blockIdx.y * 64;
    const int n0    = blockIdx.x * 128 + wave * 16;
    const bool wactive = (n0 < N);

    // cooperative A staging: thread -> (row, 8-elem col group)
    const int ar = tid >> 2;
    const int ac = (tid & 3) * 8;
    const __bf16* Arow = A + (size_t)(mbase + ar) * K;

    // B fragment addressing: lane = output column, K contiguous per lane.
    const int nl   = n0 + l15;
    const int koff = (lane < 16) ? 0 : 16;
    const __bf16* Wrow = W + (size_t)(wactive ? nl : 0) * K;

    v8f acc[4] = {v8f{}, v8f{}, v8f{}, v8f{}};

    // prologue: stage chunk 0 into buffer 0
    *(uint4*)&sA[0][ar * LDA_PAD + ac]      = *(const uint4*)&Arow[ac];
    *(uint4*)&sA[0][ar * LDA_PAD + 32 + ac] = *(const uint4*)&Arow[32 + ac];
    __syncthreads();

    int buf = 0;
    for (int k0 = 0; k0 < K; k0 += 64) {
        // --- software pipeline: pull next A chunk into registers ---
        const bool have_next = (k0 + 64) < K;
        uint4 pa0, pa1;
        if (have_next) {
            pa0 = *(const uint4*)&Arow[k0 + 64 + ac];
            pa1 = *(const uint4*)&Arow[k0 + 96 + ac];
        }

        if (wactive) {
            if (have_next) __builtin_prefetch(Wrow + k0 + 64 + koff, 0, 1);
            // both B fragments for this 64-K chunk, issued up front
            union bun { v16bf v; uint4 q[2]; };
            bun b0, b1;
            const __bf16* wp = Wrow + k0 + koff;
            b0.q[0] = *(const uint4*)(wp);
            b0.q[1] = *(const uint4*)(wp + 8);
            b1.q[0] = *(const uint4*)(wp + 32);
            b1.q[1] = *(const uint4*)(wp + 40);

            // all 8 A fragments (both 32-K sub-chunks) loaded up front ...
            const __bf16* sa = &sA[buf][0];
            union aun { v16bf v; uint4 q[2]; } af[8];
#pragma unroll
            for (int ks = 0; ks < 2; ++ks) {
                const int acol = ks * 32 + ((lane < 16) ? 0 : 16);
#pragma unroll
                for (int mt = 0; mt < 4; ++mt) {
                    const int r = mt * 16 + l15;
                    af[ks * 4 + mt].q[0] = *(const uint4*)&sa[r * LDA_PAD + acol];
                    af[ks * 4 + mt].q[1] = *(const uint4*)&sa[r * LDA_PAD + acol + 8];
                }
            }
            // ... then 8 back-to-back WMMAs (single ds wait, XDL stays busy)
            __builtin_amdgcn_sched_barrier(0);
#pragma unroll
            for (int ks = 0; ks < 2; ++ks) {
                const v16bf bb = ks ? b1.v : b0.v;
#pragma unroll
                for (int mt = 0; mt < 4; ++mt)
                    acc[mt] = __builtin_amdgcn_wmma_f32_16x16x32_bf16(
                        false, af[ks * 4 + mt].v, false, bb, (short)0, acc[mt], false, false);
            }
        }

        // --- stage next chunk into the other buffer; one barrier per chunk ---
        if (have_next) {
            const int nb = buf ^ 1;
            *(uint4*)&sA[nb][ar * LDA_PAD + ac]      = pa0;
            *(uint4*)&sA[nb][ar * LDA_PAD + 32 + ac] = pa1;
        }
        __syncthreads();
        buf ^= 1;
    }
    if (!wactive) return;

    const float bv = bias ? bias[nl] : 0.0f;
#pragma unroll
    for (int mt = 0; mt < 4; ++mt) {
#pragma unroll
        for (int j = 0; j < 8; ++j) {
            const int m = mbase + mt * 16 + ((lane < 16) ? j : j + 8);
            float v = acc[mt][j] + bv;
            if (dec_len && t >= dec_len[m]) v = 0.0f;
            if (out_bf16) ((__bf16*)out)[(size_t)m * ostride + nl] = (__bf16)v;
            else          ((float*)out)[(size_t)m * ostride + nl] = v;
        }
    }
}

// ---------------------------------------------------------------------------
// Setup kernels
// ---------------------------------------------------------------------------
__global__ void sort_prep(const int* __restrict__ cl, const int* __restrict__ caps,
                          int* __restrict__ sort_ind, int* __restrict__ dec_len,
                          int* __restrict__ caps_s, float* __restrict__ out_tail)
{
    const int i = threadIdx.x;   // 64 threads, 1 block
    const int myl = cl[i];
    int rank = 0;
    for (int j = 0; j < B_; ++j) {
        const int lj = cl[j];
        rank += (lj > myl) || (lj == myl && j < i);   // stable argsort(-cl)
    }
    sort_ind[rank] = i;
    dec_len[rank]  = myl - 1;
    for (int tt = 0; tt < ML_; ++tt) caps_s[rank * ML_ + tt] = caps[i * ML_ + tt];
    __syncthreads();
    // aux outputs (caps_sorted, dec_len, sort_ind) as floats into d_out tail
    for (int tt = 0; tt < ML_; ++tt) out_tail[i * ML_ + tt] = (float)caps_s[i * ML_ + tt];
    out_tail[B_ * ML_ + i]      = (float)dec_len[i];
    out_tail[B_ * ML_ + B_ + i] = (float)sort_ind[i];
}

__global__ void zero_f32(float* __restrict__ p, int n)
{
    for (int i = blockIdx.x * blockDim.x + threadIdx.x; i < n; i += gridDim.x * blockDim.x)
        p[i] = 0.0f;
}

__global__ void cvt_bf16(const float* __restrict__ s, __bf16* __restrict__ d, int n)
{
    for (int i = blockIdx.x * blockDim.x + threadIdx.x; i < n; i += gridDim.x * blockDim.x)
        d[i] = (__bf16)s[i];
}

// dst[r, 0:ka) = a[r,:] ; dst[r, ka:ka+kb) = b[r,:]   (K-concat of LSTM weights)
__global__ void cat_cvt_bf16(const float* __restrict__ a, int ka,
                             const float* __restrict__ b, int kb,
                             __bf16* __restrict__ dst, int rows)
{
    const int K = ka + kb;
    const long n = (long)rows * K;
    for (long i = blockIdx.x * blockDim.x + threadIdx.x; i < n; i += (long)gridDim.x * blockDim.x) {
        const int r = (int)(i / K), c = (int)(i % K);
        const float v = (c < ka) ? a[(size_t)r * ka + c] : b[(size_t)r * kb + (c - ka)];
        dst[i] = (__bf16)v;
    }
}

// Gather encoder features by sort_ind, emit bf16 copy + enc_mean into xh1 slot.
__global__ __launch_bounds__(256)
void enc_prep(const float* __restrict__ enc, const int* __restrict__ sort_ind,
              __bf16* __restrict__ enc_bf, __bf16* __restrict__ xh1)
{
    const int b = blockIdx.x;
    const int e = blockIdx.y * 256 + threadIdx.x;     // grid (64, 8)
    const int bs = sort_ind[b];
    float sum = 0.0f;
    for (int p = 0; p < P_; ++p) {
        const float v = enc[((size_t)bs * P_ + p) * ENC_ + e];
        enc_bf[((size_t)b * P_ + p) * ENC_ + e] = (__bf16)v;
        sum += v;
    }
    xh1[(size_t)b * 5120 + DEC_ + e] = (__bf16)(sum * (1.0f / (float)P_));
}

// ---------------------------------------------------------------------------
// Per-step kernels
// ---------------------------------------------------------------------------
// xh1 = [h2 | enc_mean(prefilled) | emb_t | h1]; also xh2[3072:] = old h2
__global__ void build_x(const float* __restrict__ h2s, const float* __restrict__ h1s,
                        const float* __restrict__ emb, const int* __restrict__ caps_s,
                        __bf16* __restrict__ xh1, __bf16* __restrict__ xh2, int t)
{
    const int b = blockIdx.x;
    const int tok = caps_s[b * ML_ + t];
    for (int c = threadIdx.x; c < DEC_; c += 256) {
        const float h2v = h2s[b * DEC_ + c];
        const __bf16 h2b = (__bf16)h2v;
        xh1[(size_t)b * 5120 + c]        = h2b;
        xh2[(size_t)b * 4096 + 3072 + c] = h2b;
        xh1[(size_t)b * 5120 + 4096 + c] = (__bf16)h1s[b * DEC_ + c];
        xh1[(size_t)b * 5120 + 3072 + c] = (__bf16)emb[(size_t)tok * EMB_ + c];
    }
}

__global__ void lstm_update(const float* __restrict__ gates,
                            const float* __restrict__ bih, const float* __restrict__ bhh,
                            float* __restrict__ hstate, float* __restrict__ cstate,
                            __bf16* __restrict__ hbf_a,
                            __bf16* __restrict__ hbf_b, int strideB, int offB,
                            const int* __restrict__ dec_len, int t)
{
    const int b = blockIdx.x;
    const int j = blockIdx.y * 256 + threadIdx.x;    // grid (64, 4)
    const float* g = gates + (size_t)b * 4096;
    const float gi = g[j]        + bih[j]        + bhh[j];
    const float gf = g[1024 + j] + bih[1024 + j] + bhh[1024 + j];
    const float gg = g[2048 + j] + bih[2048 + j] + bhh[2048 + j];
    const float go = g[3072 + j] + bih[3072 + j] + bhh[3072 + j];
    const float c_old = cstate[b * DEC_ + j];
    const float h_old = hstate[b * DEC_ + j];
    const float i_ = 1.0f / (1.0f + __expf(-gi));
    const float f_ = 1.0f / (1.0f + __expf(-gf));
    const float o_ = 1.0f / (1.0f + __expf(-go));
    float cn = f_ * c_old + i_ * tanhf(gg);
    float hn = o_ * tanhf(cn);
    const bool active = t < dec_len[b];
    cn = active ? cn : c_old;
    hn = active ? hn : h_old;
    cstate[b * DEC_ + j] = cn;
    hstate[b * DEC_ + j] = hn;
    const __bf16 hb = (__bf16)hn;
    hbf_a[b * DEC_ + j] = hb;
    if (hbf_b) hbf_b[(size_t)b * strideB + offB + j] = hb;
}

// Bahdanau attention fused: e = relu(att1+att2)@Wf + bf ; softmax ; awe -> xh2[0:2048)
__global__ __launch_bounds__(256)
void attention_step(const __bf16* __restrict__ att1, const float* __restrict__ att2,
                    const float* __restrict__ Wf, const float* __restrict__ bfp,
                    const __bf16* __restrict__ enc_bf, __bf16* __restrict__ xh2)
{
    __shared__ float s_att2[ATT_];
    __shared__ float s_wf[ATT_];
    __shared__ float s_e[P_ + 4];

    const int b = blockIdx.x, tid = threadIdx.x;
    const int lane = tid & 31, grp = tid >> 5;
    for (int a = tid; a < ATT_; a += 256) { s_att2[a] = att2[b * ATT_ + a]; s_wf[a] = Wf[a]; }
    __syncthreads();

    const __bf16* a1 = att1 + (size_t)b * P_ * ATT_;
    for (int p = grp; p < P_; p += 8) {
        const __bf16* row = a1 + (size_t)p * ATT_;
        float sum = 0.0f;
        for (int a = lane; a < ATT_; a += 32)
            sum += fmaxf((float)row[a] + s_att2[a], 0.0f) * s_wf[a];
        for (int off = 16; off; off >>= 1) sum += __shfl_xor(sum, off, 32);
        if (lane == 0) s_e[p] = sum + bfp[0];
    }
    __syncthreads();

    if (tid < 32) {   // softmax over 196 within one wave
        float mx = -3.4e38f;
        for (int p = lane; p < P_; p += 32) mx = fmaxf(mx, s_e[p]);
        for (int off = 16; off; off >>= 1) mx = fmaxf(mx, __shfl_xor(mx, off, 32));
        float s = 0.0f;
        for (int p = lane; p < P_; p += 32) { const float e = __expf(s_e[p] - mx); s_e[p] = e; s += e; }
        for (int off = 16; off; off >>= 1) s += __shfl_xor(s, off, 32);
        const float inv = 1.0f / s;
        for (int p = lane; p < P_; p += 32) s_e[p] *= inv;
    }
    __syncthreads();

    const int eo = tid * 8;          // 256 threads * 8 = 2048
    float acc[8] = {};
    for (int p = 0; p < P_; ++p) {
        const float ap = s_e[p];
        union { uint4 q; __bf16 h[8]; } u;
        u.q = *(const uint4*)&enc_bf[((size_t)b * P_ + p) * ENC_ + eo];
#pragma unroll
        for (int j = 0; j < 8; ++j) acc[j] += ap * (float)u.h[j];
    }
#pragma unroll
    for (int j = 0; j < 8; ++j) xh2[(size_t)b * 4096 + eo + j] = (__bf16)acc[j];
}

// ---------------------------------------------------------------------------
extern "C" void kernel_launch(void* const* d_in, const int* in_sizes, int n_in,
                              void* d_out, int out_size, void* d_ws, size_t ws_size,
                              hipStream_t stream)
{
    (void)in_sizes; (void)n_in; (void)out_size; (void)ws_size;
    const float* enc_in  = (const float*)d_in[0];
    const int*   caps_in = (const int*)  d_in[1];
    const int*   cl_in   = (const int*)  d_in[2];
    const float* emb     = (const float*)d_in[3];
    const float* We   = (const float*)d_in[4];   const float* be   = (const float*)d_in[5];
    const float* Wd   = (const float*)d_in[6];   const float* bd   = (const float*)d_in[7];
    const float* Wf   = (const float*)d_in[8];   const float* bfp  = (const float*)d_in[9];
    const float* Wih1 = (const float*)d_in[10];  const float* Whh1 = (const float*)d_in[11];
    const float* bih1 = (const float*)d_in[12];  const float* bhh1 = (const float*)d_in[13];
    const float* Wih2 = (const float*)d_in[14];  const float* Whh2 = (const float*)d_in[15];
    const float* bih2 = (const float*)d_in[16];  const float* bhh2 = (const float*)d_in[17];
    const float* Wfc1 = (const float*)d_in[18];  const float* bfc1 = (const float*)d_in[19];
    const float* Wfc  = (const float*)d_in[20];  const float* bfc  = (const float*)d_in[21];

    // ---- workspace layout (~206 MB) ----
    char* ws = (char*)d_ws;
    size_t cur = 0;
    auto alloc = [&](size_t bytes) -> void* {
        void* p = ws + cur; cur = (cur + bytes + 255) & ~(size_t)255; return p;
    };
    __bf16* W1cat   = (__bf16*)alloc((size_t)4096 * 5120 * 2);
    __bf16* W2cat   = (__bf16*)alloc((size_t)4096 * 4096 * 2);
    __bf16* Wd_bf   = (__bf16*)alloc((size_t)ATT_ * DEC_ * 2);
    __bf16* Wfc1_bf = (__bf16*)alloc((size_t)V_ * DEC_ * 2);
    __bf16* Wfc_bf  = (__bf16*)alloc((size_t)V_ * DEC_ * 2);
    __bf16* We_bf   = (__bf16*)alloc((size_t)ATT_ * ENC_ * 2);
    __bf16* enc_bf  = (__bf16*)alloc((size_t)B_ * P_ * ENC_ * 2);
    __bf16* att1_bf = (__bf16*)alloc((size_t)B_ * P_ * ATT_ * 2);
    __bf16* xh1     = (__bf16*)alloc((size_t)B_ * 5120 * 2);
    __bf16* xh2     = (__bf16*)alloc((size_t)B_ * 4096 * 2);
    __bf16* h1bf    = (__bf16*)alloc((size_t)B_ * DEC_ * 2);
    __bf16* h2bf    = (__bf16*)alloc((size_t)B_ * DEC_ * 2);
    float* gates1 = (float*)alloc((size_t)B_ * 4096 * 4);
    float* gates2 = (float*)alloc((size_t)B_ * 4096 * 4);
    float* att2f  = (float*)alloc((size_t)B_ * ATT_ * 4);
    float* h1s = (float*)alloc((size_t)B_ * DEC_ * 4);
    float* c1s = (float*)alloc((size_t)B_ * DEC_ * 4);
    float* h2s = (float*)alloc((size_t)B_ * DEC_ * 4);
    float* c2s = (float*)alloc((size_t)B_ * DEC_ * 4);
    int* sort_ind = (int*)alloc(B_ * 4);
    int* dec_len  = (int*)alloc(B_ * 4);
    int* caps_s   = (int*)alloc(B_ * ML_ * 4);

    // ---- output layout: [pred | pred1 | caps | dec_len | sort_ind] ----
    float* out = (float*)d_out;
    const size_t OPRED1 = (size_t)B_ * T_ * V_;
    const size_t OCAPS  = 2 * OPRED1;

    // ---- setup ----
    sort_prep<<<1, 64, 0, stream>>>(cl_in, caps_in, sort_ind, dec_len, caps_s, out + OCAPS);
    zero_f32<<<256, 256, 0, stream>>>(h1s, 4 * B_ * DEC_);   // h1s,c1s,h2s,c2s contiguous
    cvt_bf16<<<2048, 256, 0, stream>>>(We,   We_bf,   ATT_ * ENC_);
    cvt_bf16<<<2048, 256, 0, stream>>>(Wd,   Wd_bf,   ATT_ * DEC_);
    cvt_bf16<<<2048, 256, 0, stream>>>(Wfc1, Wfc1_bf, V_ * DEC_);
    cvt_bf16<<<2048, 256, 0, stream>>>(Wfc,  Wfc_bf,  V_ * DEC_);
    cat_cvt_bf16<<<2048, 256, 0, stream>>>(Wih1, 4096, Whh1, 1024, W1cat, 4096);
    cat_cvt_bf16<<<2048, 256, 0, stream>>>(Wih2, 3072, Whh2, 1024, W2cat, 4096);
    enc_prep<<<dim3(B_, ENC_ / 256), 256, 0, stream>>>(enc_in, sort_ind, enc_bf, xh1);

    // att1 = enc @ We.T + be  (12544 x 1024, K=2048) -> bf16
    gemm_bf16_wmma<<<dim3(ATT_ / 128, (B_ * P_) / 64), 256, 0, stream>>>(
        enc_bf, ENC_, We_bf, ATT_, be, att1_bf, ATT_, 1, nullptr, 0);

    // ---- sequential decode: 19 steps ----
    const dim3 gN4096(4096 / 128, 1), gN1024(1024 / 128, 1), gNV((V_ + 127) / 128, 1);
    for (int t = 0; t < T_; ++t) {
        build_x<<<B_, 256, 0, stream>>>(h2s, h1s, emb, caps_s, xh1, xh2, t);
        // gates1 = [h2|mean|emb|h1] @ [Wih1|Whh1]^T   (K=5120, N=4096)
        gemm_bf16_wmma<<<gN4096, 256, 0, stream>>>(
            xh1, 5120, W1cat, 4096, nullptr, gates1, 4096, 0, nullptr, 0);
        lstm_update<<<dim3(B_, 4), 256, 0, stream>>>(
            gates1, bih1, bhh1, h1s, c1s, h1bf, xh2, 4096, 2048, dec_len, t);
        // att2 = h1n @ Wd^T + bd
        gemm_bf16_wmma<<<gN1024, 256, 0, stream>>>(
            h1bf, DEC_, Wd_bf, ATT_, bd, att2f, ATT_, 0, nullptr, 0);
        // preds1 = h1n @ Wfc1^T + bfc1 (masked) -> d_out
        gemm_bf16_wmma<<<gNV, 256, 0, stream>>>(
            h1bf, DEC_, Wfc1_bf, V_, bfc1, out + OPRED1 + (size_t)t * V_,
            (long)T_ * V_, 0, dec_len, t);
        attention_step<<<B_, 256, 0, stream>>>(att1_bf, att2f, Wf, bfp, enc_bf, xh2);
        // gates2 = [awe|h1n|h2] @ [Wih2|Whh2]^T   (K=4096, N=4096)
        gemm_bf16_wmma<<<gN4096, 256, 0, stream>>>(
            xh2, 4096, W2cat, 4096, nullptr, gates2, 4096, 0, nullptr, 0);
        lstm_update<<<dim3(B_, 4), 256, 0, stream>>>(
            gates2, bih2, bhh2, h2s, c2s, h2bf, nullptr, 0, 0, dec_len, t);
        // preds = h2n @ Wfc^T + bfc (masked) -> d_out
        gemm_bf16_wmma<<<gNV, 256, 0, stream>>>(
            h2bf, DEC_, Wfc_bf, V_, bfc, out + (size_t)t * V_,
            (long)T_ * V_, 0, dec_len, t);
    }
}
